// ElasticGNN_28587302322288
// MI455X (gfx1250) — matrix operand order repaired and checked by reading, not compiled, not run on hardware
//
#include <hip/hip_runtime.h>

#define N_NODES 100000
#define IN_C    512
#define HID     256
#define OUT_C   32
#define N_EDGES 1600000
#define K_ITERS 5
#define LAM1f   3.0f
#define GAMMAf  0.25f   /* 1/(1+LAM2) */
#define BETAf   2.0f    /* 1/(2*GAMMA) */

typedef _Float16 v16h __attribute__((ext_vector_type(16)));
typedef _Float16 v8h  __attribute__((ext_vector_type(8)));
typedef float    v8f  __attribute__((ext_vector_type(8)));

static __device__ __forceinline__ void atomic_add_f32(float* p, float v) {
  __hip_atomic_fetch_add(p, v, __ATOMIC_RELAXED, __HIP_MEMORY_SCOPE_AGENT);
}

// ---------------------------------------------------------------------------
// Zero scratch that is accumulated into: z [E,32], atz [N,32], deg [N]
// ---------------------------------------------------------------------------
__global__ void prep_kernel(float* __restrict__ z, float* __restrict__ atz,
                            float* __restrict__ deg) {
  size_t t = (size_t)blockIdx.x * blockDim.x + threadIdx.x;
  if (t < (size_t)N_EDGES * 32) z[t] = 0.0f;
  if (t < (size_t)N_NODES * 32) atz[t] = 0.0f;
  if (t < (size_t)N_NODES)      deg[t] = 0.0f;
}

// ---------------------------------------------------------------------------
// One-shot weight transpose+convert to f16, K-contiguous per output channel:
//   w1t[n][k] = W1[k][n]  (n<HID,  k<IN_C)   -> B-fragment = 2x16B loads
//   w2t[n][k] = W2[k][n]  (n<OUT_C,k<HID)
// Total 0.6 MB: negligible one-time cost, saves 16 scalar loads + 16 cvts
// per WMMA in the GEMM kernels.
// ---------------------------------------------------------------------------
__global__ void convert_w_kernel(const float* __restrict__ W1,
                                 const float* __restrict__ W2,
                                 _Float16* __restrict__ w1t,
                                 _Float16* __restrict__ w2t) {
  int t = blockIdx.x * blockDim.x + threadIdx.x;
  if (t < IN_C * HID) {
    int n = t >> 9;            // t / IN_C
    int k = t & (IN_C - 1);    // t % IN_C
    w1t[t] = (_Float16)W1[(size_t)k * HID + n];
  }
  if (t < HID * OUT_C) {
    int n = t >> 8;            // t / HID
    int k = t & (HID - 1);     // t % HID
    w2t[t] = (_Float16)W2[(size_t)k * OUT_C + n];
  }
}

// ---------------------------------------------------------------------------
// Fused MLP: h2 = relu(x@W1+b1)@W2 + b2 ; also y = gamma*h2 (initial carry).
// One wave computes 16 rows. blockDim = 128 (4 waves).
// WMMA fragment layouts per CDNA5 ISA 7.12.2:
//   A 16x32 f16 : lane -> row = lane%16; halfs[0..7]=K(ks*32 + 8*(lane/16)+i),
//                 halfs[8..15]= same +16.
//   B 32x16 f16 : lane -> col = lane%16; same K mapping as A.
//   C/D 16x16 f32: lane -> col = lane%16; vgpr v -> row = v + 8*(lane/16).
// ---------------------------------------------------------------------------
__global__ __launch_bounds__(128) void fused_mlp_kernel(
    const float* __restrict__ x,
    const _Float16* __restrict__ w1t, const float* __restrict__ b1,
    const _Float16* __restrict__ w2t, const float* __restrict__ b2,
    float* __restrict__ h2, float* __restrict__ y)
{
  __shared__ _Float16 sh1[4][16][HID + 8];   // h1 tile per wave, padded rows

  const int wave = threadIdx.x >> 5;
  const int lane = threadIdx.x & 31;
  const int lmod = lane & 15;
  const int hsel = lane >> 4;                 // 0 or 1

  int tile = blockIdx.x * 4 + wave;
  const bool active = tile < (N_NODES / 16);
  if (!active) tile = (N_NODES / 16) - 1;     // keep full-wave EXEC for WMMA
  const int row0 = tile * 16;

  // ---------------- GEMM1: h1 = relu(x @ W1 + b1) -> LDS (f16) -------------
  const float* xrow = x + (size_t)(row0 + lmod) * IN_C;
  for (int j = 0; j < HID / 16; ++j) {
    const int col = j * 16 + lmod;
    const float binit = b1[col];
    v8f acc;
#pragma unroll
    for (int v = 0; v < 8; ++v) acc[v] = binit;

    const _Float16* w1row = w1t + (size_t)col * IN_C + hsel * 8;

    for (int ks = 0; ks < IN_C / 32; ++ks) {
      // A fragment: on-the-fly f32 -> f16 (x read once, at its HBM floor)
      const float* ap = xrow + ks * 32 + hsel * 8;
      float4 a0 = *(const float4*)(ap);
      float4 a1 = *(const float4*)(ap + 4);
      float4 a2 = *(const float4*)(ap + 16);
      float4 a3 = *(const float4*)(ap + 20);
      v16h A;
      A[0]=(_Float16)a0.x;  A[1]=(_Float16)a0.y;  A[2]=(_Float16)a0.z;  A[3]=(_Float16)a0.w;
      A[4]=(_Float16)a1.x;  A[5]=(_Float16)a1.y;  A[6]=(_Float16)a1.z;  A[7]=(_Float16)a1.w;
      A[8]=(_Float16)a2.x;  A[9]=(_Float16)a2.y;  A[10]=(_Float16)a2.z; A[11]=(_Float16)a2.w;
      A[12]=(_Float16)a3.x; A[13]=(_Float16)a3.y; A[14]=(_Float16)a3.z; A[15]=(_Float16)a3.w;

      // B fragment: two runs of 8 K halves (+0, +16) -> two 16B vector loads
      const _Float16* bp = w1row + ks * 32;
      v8h blo = *(const v8h*)(bp);
      v8h bhi = *(const v8h*)(bp + 16);
      v16h B;
#pragma unroll
      for (int h = 0; h < 8; ++h) { B[h] = blo[h]; B[8 + h] = bhi[h]; }

      acc = __builtin_amdgcn_wmma_f32_16x16x32_f16(false, A, false, B,
                                                   (short)0, acc, false, false);
    }
#pragma unroll
    for (int v = 0; v < 8; ++v) {
      float r = acc[v];
      r = r > 0.0f ? r : 0.0f;                         // ReLU
      sh1[wave][hsel * 8 + v][col] = (_Float16)r;      // f16 for GEMM2
    }
  }
  // LDS ops from one wave complete in order (DScnt in-order) — no barrier
  // needed for this intra-wave handoff.

  // ---------------- GEMM2: h2 = h1 @ W2 + b2 -------------------------------
  for (int n2 = 0; n2 < OUT_C / 16; ++n2) {
    const int col = n2 * 16 + lmod;
    const float binit = b2[col];
    v8f acc;
#pragma unroll
    for (int v = 0; v < 8; ++v) acc[v] = binit;

    const _Float16* w2row = w2t + (size_t)col * HID + hsel * 8;

    for (int ks = 0; ks < HID / 32; ++ks) {
      // A from LDS: two aligned 16B ds loads (row stride 528B = 33*16B)
      const _Float16* ap = &sh1[wave][lmod][ks * 32 + hsel * 8];
      v8h alo = *(const v8h*)(ap);
      v8h ahi = *(const v8h*)(ap + 16);
      v16h A;
#pragma unroll
      for (int h = 0; h < 8; ++h) { A[h] = alo[h]; A[8 + h] = ahi[h]; }

      const _Float16* bp = w2row + ks * 32;
      v8h blo = *(const v8h*)(bp);
      v8h bhi = *(const v8h*)(bp + 16);
      v16h B;
#pragma unroll
      for (int h = 0; h < 8; ++h) { B[h] = blo[h]; B[8 + h] = bhi[h]; }

      acc = __builtin_amdgcn_wmma_f32_16x16x32_f16(false, A, false, B,
                                                   (short)0, acc, false, false);
    }
    if (active) {
#pragma unroll
      for (int v = 0; v < 8; ++v) {
        const size_t idx = (size_t)(row0 + hsel * 8 + v) * OUT_C + col;
        h2[idx] = acc[v];
        y[idx]  = GAMMAf * acc[v];      // y0 = gamma * h
      }
    }
  }
}

// ---------------------------------------------------------------------------
// Degree accumulation:  deg[row]+=1 ; deg[col]+=1   (exact in f32)
// ---------------------------------------------------------------------------
__global__ void deg_kernel(const long long* __restrict__ ei, float* __restrict__ deg) {
  size_t e = (size_t)blockIdx.x * blockDim.x + threadIdx.x;
  if (e >= (size_t)N_EDGES) return;
  int r = (int)ei[e];
  int c = (int)ei[(size_t)N_EDGES + e];
  atomic_add_f32(deg + r, 1.0f);
  atomic_add_f32(deg + c, 1.0f);
}

__global__ void dis_kernel(const float* __restrict__ deg, float* __restrict__ dis) {
  size_t n = (size_t)blockIdx.x * blockDim.x + threadIdx.x;
  if (n >= (size_t)N_NODES) return;
  dis[n] = rsqrtf(fmaxf(deg[n], 1.0f));
}

// ---------------------------------------------------------------------------
// Edge step: z = clip(z + beta*(y[row]*dr - y[col]*dc), +/-lam1);
//            atz[row] += z*dr ; atz[col] -= z*dc
// lane = channel (32 channels == wave32): z fully coalesced, edge-index loads
// wave-uniform, y/atz (12.8 MB each) are L2-resident on the 192 MB L2.
// ---------------------------------------------------------------------------
__global__ void edge_kernel(const long long* __restrict__ ei,
                            const float* __restrict__ dis,
                            const float* __restrict__ y,
                            float* __restrict__ z,
                            float* __restrict__ atz)
{
  size_t t = (size_t)blockIdx.x * blockDim.x + threadIdx.x;   // = e*32 + c
  size_t e = t >> 5;
  int    c = (int)(t & 31);
  if (e >= (size_t)N_EDGES) return;
  int r  = (int)ei[e];
  int cl = (int)ei[(size_t)N_EDGES + e];
  float dr = dis[r];
  float dc = dis[cl];
  float zn = z[t] + BETAf * (y[(size_t)r * 32 + c] * dr - y[(size_t)cl * 32 + c] * dc);
  zn = fminf(fmaxf(zn, -LAM1f), LAM1f);
  z[t] = zn;
  atomic_add_f32(&atz[(size_t)r  * 32 + c],  zn * dr);
  atomic_add_f32(&atz[(size_t)cl * 32 + c], -zn * dc);
}

// ---------------------------------------------------------------------------
// Node step: y = gamma*h - gamma*atz ; atz = 0 (for next iteration).
// On the final iteration also emit log_softmax over the 32 channels, which
// map exactly onto one wave32 -> pure __shfl_xor reductions.
// ---------------------------------------------------------------------------
__global__ void node_kernel(const float* __restrict__ h2,
                            float* __restrict__ atz,
                            float* __restrict__ y,
                            float* __restrict__ out,
                            int do_out)
{
  size_t t = (size_t)blockIdx.x * blockDim.x + threadIdx.x;   // = n*32 + c
  if (t >= (size_t)N_NODES * 32) return;
  float a  = atz[t];
  float yv = GAMMAf * h2[t] - GAMMAf * a;
  atz[t] = 0.0f;
  y[t] = yv;
  if (do_out) {
    float m = yv;
#pragma unroll
    for (int o = 16; o > 0; o >>= 1) m = fmaxf(m, __shfl_xor(m, o, 32));
    float ex = expf(yv - m);
    float s = ex;
#pragma unroll
    for (int o = 16; o > 0; o >>= 1) s += __shfl_xor(s, o, 32);
    out[t] = yv - m - logf(s);
  }
}

// ---------------------------------------------------------------------------
extern "C" void kernel_launch(void* const* d_in, const int* in_sizes, int n_in,
                              void* d_out, int out_size, void* d_ws, size_t ws_size,
                              hipStream_t stream) {
  const float*     x  = (const float*)d_in[0];
  const long long* ei = (const long long*)d_in[1];
  const float*     W1 = (const float*)d_in[2];
  const float*     b1 = (const float*)d_in[3];
  const float*     W2 = (const float*)d_in[4];
  const float*     b2 = (const float*)d_in[5];
  float* out = (float*)d_out;

  // Workspace carve-out (256 B aligned): ~233 MiB total
  char* ws = (char*)d_ws;
  size_t off = 0;
  auto carve = [&](size_t bytes) -> char* {
    char* p = ws + off;
    off += (bytes + 255) & ~(size_t)255;
    return p;
  };
  float* z    = (float*)carve((size_t)N_EDGES * 32 * sizeof(float));  // 204.8 MB
  float* atz  = (float*)carve((size_t)N_NODES * 32 * sizeof(float));  // 12.8 MB
  float* y    = (float*)carve((size_t)N_NODES * 32 * sizeof(float));  // 12.8 MB
  float* h2   = (float*)carve((size_t)N_NODES * 32 * sizeof(float));  // 12.8 MB
  float* deg  = (float*)carve((size_t)N_NODES * sizeof(float));
  float* dis  = (float*)carve((size_t)N_NODES * sizeof(float));
  _Float16* w1t = (_Float16*)carve((size_t)IN_C * HID * sizeof(_Float16));  // 256 KB
  _Float16* w2t = (_Float16*)carve((size_t)HID * OUT_C * sizeof(_Float16)); // 16 KB

  const int ethreads  = N_EDGES * 32;                 // 51.2M
  const int eblocks   = (ethreads + 255) / 256;       // 200000
  const int nblocks32 = (N_NODES * 32 + 255) / 256;   // 12500

  prep_kernel<<<eblocks, 256, 0, stream>>>(z, atz, deg);
  convert_w_kernel<<<(IN_C * HID + 255) / 256, 256, 0, stream>>>(W1, W2, w1t, w2t);

  const int tiles = N_NODES / 16;                     // 6250
  fused_mlp_kernel<<<(tiles + 3) / 4, 128, 0, stream>>>(x, w1t, b1, w2t, b2, h2, y);

  deg_kernel<<<(N_EDGES + 255) / 256, 256, 0, stream>>>(ei, deg);
  dis_kernel<<<(N_NODES + 255) / 256, 256, 0, stream>>>(deg, dis);

  for (int k = 0; k < K_ITERS; ++k) {
    edge_kernel<<<eblocks, 256, 0, stream>>>(ei, dis, y, z, atz);
    node_kernel<<<nblocks32, 256, 0, stream>>>(h2, atz, y, out, k == K_ITERS - 1 ? 1 : 0);
  }
}
